// SwinTransformerBlock_48524540510917
// MI455X (gfx1250) — matrix operand order repaired and checked by reading, not compiled, not run on hardware
//
#include <hip/hip_runtime.h>
#include <math.h>

// ---------- problem constants ----------
#define DIM    512
#define HEADS  16
#define HD     32
#define WS     7
#define NTOK   49          // tokens per window
#define RB     8
#define NWIN   64
#define BWIN   (RB*NWIN)   // 512 windows
#define NRF    16
#define MROWS  (BWIN*NTOK) // 25088
#define SLICE  (NWIN*NTOK) // 3136 conv "height"
#define SLICEE (SLICE*16)  // 50176 elems per (rb,h)
#define SCALEF 0.17677669529663687f  // 32^-0.5
#define LNEPS  1e-5f

typedef float v8f  __attribute__((ext_vector_type(8)));
typedef __bf16 v16bf __attribute__((ext_vector_type(16)));

union BF16Frag { v16bf v; unsigned u[8]; unsigned short h[16]; };

__device__ __forceinline__ unsigned short f2bf(float f) {
  unsigned u = __float_as_uint(f);
  unsigned r = u + 0x7FFFu + ((u >> 16) & 1u);   // RNE
  return (unsigned short)(r >> 16);
}

__device__ __forceinline__ v8f wmma_bf16(const BF16Frag& a, const BF16Frag& b, v8f c) {
  return __builtin_amdgcn_wmma_f32_16x16x32_bf16(false, a.v, false, b.v, (short)0, c, false, false);
}

// CDNA5 async DMA: copy 16 bytes/lane global -> LDS, tracked by ASYNCcnt.
__device__ __forceinline__ void async_copy16(unsigned short* lds_dst, const unsigned short* gsrc) {
  unsigned loff = (unsigned)(unsigned long long)lds_dst;   // LDS aperture: addr[31:0] = LDS offset
  asm volatile("global_load_async_to_lds_b128 %0, %1, off"
               :: "v"(loff), "v"(gsrc)
               : "memory");
}
__device__ __forceinline__ void wait_async0() {
  asm volatile("s_wait_asynccnt 0x0" ::: "memory");
}

// A-fragment (16x32 bf16, MxK) from a row-major [rows x 32] bf16 buffer
__device__ __forceinline__ void loadA32(const unsigned short* S, int mt, int lane, BF16Frag& f) {
  int m  = mt * 16 + (lane & 15);
  int kb = (lane < 16) ? 0 : 16;
#pragma unroll
  for (int vv = 0; vv < 8; ++vv) {
    int kk = kb + (vv << 1);
    f.u[vv] = *(const unsigned*)&S[m * 32 + kk];
  }
}
// A-fragment from a row-major [64 x 64] bf16 buffer, K chunk ks (0/1)
__device__ __forceinline__ void loadA64(const unsigned short* S, int mt, int ks, int lane, BF16Frag& f) {
  int m  = mt * 16 + (lane & 15);
  int kb = ((lane < 16) ? 0 : 16) + ks * 32;
#pragma unroll
  for (int vv = 0; vv < 8; ++vv) {
    int kk = kb + (vv << 1);
    f.u[vv] = *(const unsigned*)&S[m * 64 + kk];
  }
}

// ---------------- fp32 -> bf16 conversion ----------------
__global__ void f32_to_bf16_kernel(const float* __restrict__ in, unsigned short* __restrict__ out, int n) {
  int i = blockIdx.x * 256 + threadIdx.x;
  if (i < n) out[i] = f2bf(in[i]);
}

// ---------------- 64x64 WMMA GEMM with async-DMA double-buffered staging ----------------
// C = A(MxK, bf16) * W(NxK, bf16)^T + bias
// EPI==0: QKV scatter into q/k/v bf16 (b,h,tok,hd), q scaled by SCALEF
// EPI==1: fp32 row-major store into out0
template<int EPI>
__global__ __launch_bounds__(256) void gemm64x64_kernel(
    const unsigned short* __restrict__ Abf, const unsigned short* __restrict__ Wbf,
    const float* __restrict__ bias,
    void* out0, void* out1, void* out2, int M, int N, int K) {
  __shared__ __align__(16) unsigned short sA[2][64 * 32];
  __shared__ __align__(16) unsigned short sB[2][64 * 32];
  const int tid  = threadIdx.x;
  const int lane = tid & 31;
  const int wv   = tid >> 5;
  const int nt   = wv & 3;
  const int mt0  = (wv >> 2) * 2;
  const long mBase = (long)blockIdx.x * 64;
  const long nBase = (long)blockIdx.y * 64;

  // per-thread 16B copy slot: row = tid/4, col = (tid%4)*8 bf16 elems
  const int crow = tid >> 2;
  const int ccol = (tid & 3) * 8;

  v8f c0 = {}; v8f c1 = {};
  const int nsteps = K / 32;

  // prologue: fill buffer 0
  async_copy16(&sA[0][crow * 32 + ccol], &Abf[(mBase + crow) * (long)K + ccol]);
  async_copy16(&sB[0][crow * 32 + ccol], &Wbf[(nBase + crow) * (long)K + ccol]);
  wait_async0();
  __syncthreads();

  for (int s = 0; s < nsteps; ++s) {
    const int buf = s & 1;
    if (s + 1 < nsteps) {
      int k1 = (s + 1) * 32;
      async_copy16(&sA[buf ^ 1][crow * 32 + ccol], &Abf[(mBase + crow) * (long)K + k1 + ccol]);
      async_copy16(&sB[buf ^ 1][crow * 32 + ccol], &Wbf[(nBase + crow) * (long)K + k1 + ccol]);
    }
    // B fragment (K=32 x N=16) for this wave's n-tile
    BF16Frag bfr;
    {
      int nc = nt * 16 + (lane & 15);
      int kb = (lane < 16) ? 0 : 16;
#pragma unroll
      for (int vv = 0; vv < 8; ++vv)
        bfr.u[vv] = *(const unsigned*)&sB[buf][nc * 32 + kb + 2 * vv];
    }
    BF16Frag a0, a1;
    loadA32(sA[buf], mt0, lane, a0);
    loadA32(sA[buf], mt0 + 1, lane, a1);
    c0 = wmma_bf16(a0, bfr, c0);
    c1 = wmma_bf16(a1, bfr, c1);
    if (s + 1 < nsteps) wait_async0();
    __syncthreads();
  }

  const int half = (lane >= 16) ? 8 : 0;
  const int ncol = (int)nBase + nt * 16 + (lane & 15);
#pragma unroll
  for (int t = 0; t < 2; ++t) {
    v8f c = t ? c1 : c0;
    int mt = mt0 + t;
#pragma unroll
    for (int vv = 0; vv < 8; ++vv) {
      long m = mBase + mt * 16 + vv + half;
      float val = c[vv] + bias[ncol];
      if constexpr (EPI == 0) {
        int part = ncol / DIM;
        int rem  = ncol - part * DIM;
        int hh = rem >> 5, dd = rem & 31;
        long b = m / NTOK, tok = m - b * NTOK;
        long di = ((b * HEADS + hh) * NTOK + tok) * HD + dd;
        unsigned short bv = f2bf(part == 0 ? val * SCALEF : val);
        if (part == 0)      ((unsigned short*)out0)[di] = bv;
        else if (part == 1) ((unsigned short*)out1)[di] = bv;
        else                ((unsigned short*)out2)[di] = bv;
      } else {
        ((float*)out0)[m * (long)N + ncol] = val;
      }
    }
  }
}

// ---------------- reference-token projection (tiny, fp32) ----------------
__global__ __launch_bounds__(256) void ref_proj_kernel(
    const float* __restrict__ x_ref, const float* __restrict__ refqk_w,
    const float* __restrict__ refqk_b, const float* __restrict__ dmu,
    const float* __restrict__ dls,
    unsigned short* __restrict__ refq_bf, unsigned short* __restrict__ refv_bf) {
  int idx = blockIdx.x * 256 + threadIdx.x;      // 128*1024
  int row = idx >> 10, col = idx & 1023;
  const float* xr = x_ref + row * DIM;
  const float* wr = refqk_w + (long)col * DIM;
  float acc = refqk_b[col];
  for (int k = 0; k < DIM; ++k) acc += xr[k] * wr[k];
  int rb = row >> 4, t = row & 15;
  if (col < DIM) {
    float val = dmu[col] + expf(dls[col]) * acc;
    int h = col >> 5, d = col & 31;
    refq_bf[((rb * HEADS + h) * NRF + t) * HD + d] = f2bf(val);
  } else {
    int c = col - DIM;
    int h = c >> 5, d = c & 31;
    refv_bf[((rb * HEADS + h) * NRF + t) * HD + d] = f2bf(acc);
  }
}

// ---------------- ref_attn = q(49x32) . ref_k(16x32)^T per (b,h) ----------------
__global__ __launch_bounds__(256) void ref_attn_kernel(
    const unsigned short* __restrict__ q_bf, const unsigned short* __restrict__ refq_bf,
    float* __restrict__ ra) {
  int g = blockIdx.x * 8 + (threadIdx.x >> 5);   // 0..8191
  int lane = threadIdx.x & 31;
  int b = g >> 4, h = g & 15;
  int rb = b >> 6, win = b & 63;
  BF16Frag bf;
  {
    int r = lane & 15;
    int kb = (lane < 16) ? 0 : 16;
    const unsigned short* base = refq_bf + ((rb * HEADS + h) * NRF + r) * HD;
#pragma unroll
    for (int vv = 0; vv < 8; ++vv)
      bf.u[vv] = *(const unsigned*)&base[kb + 2 * vv];
  }
  const unsigned short* qb = q_bf + (long)(b * HEADS + h) * NTOK * HD;
  float* rbase = ra + ((long)(rb * HEADS + h) * SLICE + win * NTOK) * NRF;
  const int half = (lane >= 16) ? 8 : 0;
#pragma unroll
  for (int mt = 0; mt < 4; ++mt) {
    BF16Frag af;
    int m = mt * 16 + (lane & 15);
    int kb = (lane < 16) ? 0 : 16;
    if (m < NTOK) {
#pragma unroll
      for (int vv = 0; vv < 8; ++vv) {
        int kk = kb + (vv << 1);
        af.u[vv] = *(const unsigned*)&qb[m * HD + kk];
      }
    } else {
#pragma unroll
      for (int vv = 0; vv < 8; ++vv) af.u[vv] = 0;
    }
    v8f c = {};
    c = wmma_bf16(af, bf, c);
    int r = lane & 15;
#pragma unroll
    for (int vv = 0; vv < 8; ++vv) {
      int tok = mt * 16 + vv + half;
      if (tok < NTOK) rbase[tok * NRF + r] = c[vv];
    }
  }
}

// ---------------- diffusion conv block (fp32) ----------------
__global__ void zero_stats_kernel(float* stats) {
  int i = blockIdx.x * 256 + threadIdx.x;
  if (i < 256) stats[i] = 0.0f;
}

__global__ __launch_bounds__(256) void conv_kernel(
    const float* __restrict__ ra, const float* __restrict__ conv_w,
    const float* __restrict__ conv_b, float* __restrict__ u, float* __restrict__ stats) {
  int slice = blockIdx.y;                 // rb*16 + cout
  int rb = slice >> 4, co = slice & 15;
  __shared__ float w[144];
  __shared__ float p1[8], p2[8];
  int tid = threadIdx.x;
  if (tid < 144) w[tid] = conv_w[co * 144 + tid];
  __syncthreads();
  int e = blockIdx.x * 256 + tid;
  int y = e >> 4, xw = e & 15;
  const float* rbase = ra + (long)rb * HEADS * SLICEE;
  float acc = conv_b[co];
#pragma unroll 4
  for (int ci = 0; ci < 16; ++ci) {
    const float* ib = rbase + (long)ci * SLICEE;
    const float* wc = w + ci * 9;
    for (int dy = -1; dy <= 1; ++dy) {
      int yy = y + dy;
      if (yy < 0 || yy >= SLICE) continue;
      for (int dx = -1; dx <= 1; ++dx) {
        int xx = xw + dx;
        if (xx < 0 || xx >= 16) continue;
        acc += ib[yy * 16 + xx] * wc[(dy + 1) * 3 + (dx + 1)];
      }
    }
  }
  u[(long)slice * SLICEE + e] = acc;
  float s1 = acc, s2 = acc * acc;
  for (int m = 16; m >= 1; m >>= 1) {
    s1 += __shfl_xor(s1, m, 32);
    s2 += __shfl_xor(s2, m, 32);
  }
  int wv = tid >> 5, ln = tid & 31;
  if (ln == 0) { p1[wv] = s1; p2[wv] = s2; }
  __syncthreads();
  if (tid == 0) {
    float t1 = 0.f, t2 = 0.f;
    for (int i = 0; i < 8; ++i) { t1 += p1[i]; t2 += p2[i]; }
    atomicAdd(&stats[slice * 2], t1);
    atomicAdd(&stats[slice * 2 + 1], t2);
  }
}

__global__ __launch_bounds__(256) void ln_gelu_kernel(
    const float* __restrict__ u, const float* __restrict__ stats, float* __restrict__ ra) {
  int slice = blockIdx.y;
  int e = blockIdx.x * 256 + threadIdx.x;
  const float inv = 1.0f / (float)SLICEE;
  float mean = stats[slice * 2] * inv;
  float var  = stats[slice * 2 + 1] * inv - mean * mean;
  float rstd = rsqrtf(var + LNEPS);
  long idx = (long)slice * SLICEE + e;
  float un = (u[idx] - mean) * rstd;
  float g = 0.5f * un * (1.0f + erff(un * 0.70710678118654752f));
  ra[idx] += g;
}

// ---------------- fused per-window attention (one wave per (b,h)) ----------------
__global__ __launch_bounds__(128) void attn_kernel(
    const float* __restrict__ ra, const unsigned short* __restrict__ k_bf,
    const unsigned short* __restrict__ v_bf, const unsigned short* __restrict__ refv_bf,
    const float* __restrict__ rpb, const int* __restrict__ rel_idx,
    unsigned short* __restrict__ O_bf) {
  __shared__ unsigned short sPQ[4][64 * 32];   // P (softmax probs), then reused per-row for Q
  __shared__ unsigned short sPR[4][64 * 64];   // attention probabilities
  const int wv = threadIdx.x >> 5;
  const int lane = threadIdx.x & 31;
  unsigned short* P  = sPQ[wv];
  unsigned short* PR = sPR[wv];
  int g = blockIdx.x * 4 + wv;                 // 0..8191
  int b = g >> 4, h = g & 15;
  int rb = b >> 6, win = b & 63;
  const int half = (lane >= 16) ? 8 : 0;
  const int lc = lane & 15;

  // ---- step a: row softmax of ra (49 rows x 16) -> P (64x32 bf16, zero-padded) ----
  for (int i = lane; i < 64 * 16; i += 32) ((unsigned*)P)[i] = 0;
  const float* rbase = ra + ((long)(rb * HEADS + h) * SLICE + win * NTOK) * NRF;
  for (int row = lane; row < NTOK; row += 32) {
    float vals[16]; float mx = -1e30f;
#pragma unroll
    for (int j = 0; j < 16; ++j) { vals[j] = rbase[row * NRF + j]; mx = fmaxf(mx, vals[j]); }
    float s = 0.f;
#pragma unroll
    for (int j = 0; j < 16; ++j) { vals[j] = expf(vals[j] - mx); s += vals[j]; }
    float is = 1.0f / s;
#pragma unroll
    for (int j = 0; j < 16; ++j) P[row * 32 + j] = f2bf(vals[j] * is);
  }

  // ---- step b: q_new = P(49x16) @ ref_v(16x32) * SCALE, store into Q (reuse P rows) ----
  BF16Frag bv[2];
  const unsigned short* refv = refv_bf + (long)(rb * HEADS + h) * NRF * HD;
#pragma unroll
  for (int jt = 0; jt < 2; ++jt) {
    if (lane < 16) {
      int d = jt * 16 + lc;
#pragma unroll
      for (int vv = 0; vv < 8; ++vv) {
        unsigned lo = refv[(2 * vv) * HD + d];
        unsigned hi = refv[(2 * vv + 1) * HD + d];
        bv[jt].u[vv] = lo | (hi << 16);
      }
    } else {
#pragma unroll
      for (int vv = 0; vv < 8; ++vv) bv[jt].u[vv] = 0;   // K rows 16..31 are zero
    }
  }
#pragma unroll
  for (int mt = 0; mt < 4; ++mt) {
    BF16Frag af;
    loadA32(P, mt, lane, af);
#pragma unroll
    for (int jt = 0; jt < 2; ++jt) {
      v8f c = {};
      c = wmma_bf16(af, bv[jt], c);
      int col = jt * 16 + lc;
#pragma unroll
      for (int vv = 0; vv < 8; ++vv) {
        int rrow = mt * 16 + vv + half;
        P[rrow * 32 + col] = f2bf(c[vv] * SCALEF);       // Q overwrites consumed P rows
      }
    }
  }

  // ---- step c: attn = Q(49x32) @ k^T (+rel-pos bias), row softmax -> PR (bf16) ----
  const unsigned short* kb = k_bf + (long)(b * HEADS + h) * NTOK * HD;
  const unsigned short* vb = v_bf + (long)(b * HEADS + h) * NTOK * HD;
  BF16Frag bk[4];
#pragma unroll
  for (int nt = 0; nt < 4; ++nt) {
    int m = nt * 16 + lc;
    int kbs = (lane < 16) ? 0 : 16;
    if (m < NTOK) {
#pragma unroll
      for (int vv = 0; vv < 8; ++vv)
        bk[nt].u[vv] = *(const unsigned*)&kb[m * HD + kbs + 2 * vv];
    } else {
#pragma unroll
      for (int vv = 0; vv < 8; ++vv) bk[nt].u[vv] = 0;
    }
  }
#pragma unroll
  for (int mt = 0; mt < 4; ++mt) {
    BF16Frag aq;
    loadA32(P, mt, lane, aq);
    v8f acc[4];
#pragma unroll
    for (int nt = 0; nt < 4; ++nt) { acc[nt] = {}; acc[nt] = wmma_bf16(aq, bk[nt], acc[nt]); }
#pragma unroll
    for (int vv = 0; vv < 8; ++vv) {
      int n = mt * 16 + vv + half;
      float vals[4];
#pragma unroll
      for (int nt = 0; nt < 4; ++nt) {
        int m = nt * 16 + lc;
        float x;
        if (n < NTOK && m < NTOK) x = acc[nt][vv] + rpb[rel_idx[n * NTOK + m] * HEADS + h];
        else x = -1e30f;
        vals[nt] = x;
      }
      float mx = fmaxf(fmaxf(vals[0], vals[1]), fmaxf(vals[2], vals[3]));
      for (int sh = 1; sh < 16; sh <<= 1) mx = fmaxf(mx, __shfl_xor(mx, sh, 32));
      float s = 0.f;
#pragma unroll
      for (int nt = 0; nt < 4; ++nt) {
        vals[nt] = (vals[nt] > -1e29f) ? expf(vals[nt] - mx) : 0.0f;
        s += vals[nt];
      }
      for (int sh = 1; sh < 16; sh <<= 1) s += __shfl_xor(s, sh, 32);
      float is = (s > 0.f) ? (1.0f / s) : 0.0f;
#pragma unroll
      for (int nt = 0; nt < 4; ++nt) PR[n * 64 + nt * 16 + lc] = f2bf(vals[nt] * is);
    }
  }

  // ---- step d: out = prob(49x49) @ v(49x32) -> O bf16 ----
#pragma unroll
  for (int mt = 0; mt < 4; ++mt) {
    v8f o[2] = { {}, {} };
#pragma unroll
    for (int ks = 0; ks < 2; ++ks) {
      BF16Frag ap;
      loadA64(PR, mt, ks, lane, ap);
#pragma unroll
      for (int jt = 0; jt < 2; ++jt) {
        BF16Frag bvf;
        int d = jt * 16 + lc;
        int kbs = ((lane < 16) ? 0 : 16) + ks * 32;
#pragma unroll
        for (int vv = 0; vv < 8; ++vv) {
          int m0 = kbs + 2 * vv, m1 = m0 + 1;
          unsigned lo = (m0 < NTOK) ? vb[m0 * HD + d] : 0u;
          unsigned hi = (m1 < NTOK) ? vb[m1 * HD + d] : 0u;
          bvf.u[vv] = lo | (hi << 16);
        }
        o[jt] = wmma_bf16(ap, bvf, o[jt]);
      }
    }
#pragma unroll
    for (int jt = 0; jt < 2; ++jt) {
#pragma unroll
      for (int vv = 0; vv < 8; ++vv) {
        int tok = mt * 16 + vv + half;
        if (tok < NTOK)
          O_bf[((long)b * NTOK + tok) * DIM + h * HD + jt * 16 + lc] = f2bf(o[jt][vv]);
      }
    }
  }
}

// ---------------- host side ----------------
extern "C" void kernel_launch(void* const* d_in, const int* in_sizes, int n_in,
                              void* d_out, int out_size, void* d_ws, size_t ws_size,
                              hipStream_t stream) {
  const float* x       = (const float*)d_in[0];
  const float* x_ref   = (const float*)d_in[1];
  const float* qkv_w   = (const float*)d_in[2];
  const float* qkv_b   = (const float*)d_in[3];
  const float* refqk_w = (const float*)d_in[4];
  const float* refqk_b = (const float*)d_in[5];
  const float* proj_w  = (const float*)d_in[6];
  const float* proj_b  = (const float*)d_in[7];
  const float* rpb     = (const float*)d_in[8];
  const float* dmu     = (const float*)d_in[9];
  const float* dls     = (const float*)d_in[10];
  const float* conv_w  = (const float*)d_in[11];
  const float* conv_b  = (const float*)d_in[12];
  const int*   rel_idx = (const int*)d_in[13];
  float* out = (float*)d_out;

  char* p = (char*)d_ws;
  auto alloc = [&](size_t bytes) { char* r = p; p += (bytes + 255) & ~(size_t)255; return r; };

  const size_t XN = (size_t)MROWS * DIM;                        // 12,845,056
  const size_t QKV_ELEM = (size_t)BWIN * HEADS * NTOK * HD;     // 12,845,056
  unsigned short* x_bf    = (unsigned short*)alloc(XN * 2);
  unsigned short* qkvw_bf = (unsigned short*)alloc((size_t)3 * DIM * DIM * 2);
  unsigned short* projw_bf = (unsigned short*)alloc((size_t)DIM * DIM * 2);
  unsigned short* q_bf = (unsigned short*)alloc(QKV_ELEM * 2);
  unsigned short* k_bf = (unsigned short*)alloc(QKV_ELEM * 2);
  unsigned short* v_bf = (unsigned short*)alloc(QKV_ELEM * 2);
  unsigned short* refq_bf = (unsigned short*)alloc((size_t)RB * HEADS * NRF * HD * 2);
  unsigned short* refv_bf = (unsigned short*)alloc((size_t)RB * HEADS * NRF * HD * 2);
  float* ra    = (float*)alloc((size_t)RB * HEADS * SLICEE * 4);
  float* u     = (float*)alloc((size_t)RB * HEADS * SLICEE * 4);
  float* stats = (float*)alloc(256 * 4);
  unsigned short* O_bf = (unsigned short*)alloc((size_t)MROWS * DIM * 2);

  // 1) activations + weights -> bf16
  f32_to_bf16_kernel<<<(int)((XN + 255) / 256), 256, 0, stream>>>(x, x_bf, (int)XN);
  f32_to_bf16_kernel<<<(3 * DIM * DIM + 255) / 256, 256, 0, stream>>>(qkv_w, qkvw_bf, 3 * DIM * DIM);
  f32_to_bf16_kernel<<<(DIM * DIM + 255) / 256, 256, 0, stream>>>(proj_w, projw_bf, DIM * DIM);

  // 2) QKV GEMM: (25088x512)x(512x1536), async-DMA staged
  gemm64x64_kernel<0><<<dim3(MROWS / 64, (3 * DIM) / 64), 256, 0, stream>>>(
      x_bf, qkvw_bf, qkv_b, q_bf, k_bf, v_bf, MROWS, 3 * DIM, DIM);

  // 3) reference projection (+ reparameterization)
  ref_proj_kernel<<<(RB * NRF * 2 * DIM) / 256, 256, 0, stream>>>(
      x_ref, refqk_w, refqk_b, dmu, dls, refq_bf, refv_bf);

  // 4) ref_attn = q . ref_k^T  -> ra (conv layout)
  ref_attn_kernel<<<(BWIN * HEADS) / 8, 256, 0, stream>>>(q_bf, refq_bf, ra);

  // 5) 3x diffusion conv + LN + GELU residual
  for (int it = 0; it < 3; ++it) {
    zero_stats_kernel<<<1, 256, 0, stream>>>(stats);
    conv_kernel<<<dim3(SLICEE / 256, RB * HEADS), 256, 0, stream>>>(ra, conv_w, conv_b, u, stats);
    ln_gelu_kernel<<<dim3(SLICEE / 256, RB * HEADS), 256, 0, stream>>>(u, stats, ra);
  }

  // 6) fused per-window attention
  attn_kernel<<<(BWIN * HEADS) / 4, 128, 0, stream>>>(ra, k_bf, v_bf, refv_bf, rpb, rel_idx, O_bf);

  // 7) projection GEMM: (25088x512)x(512x512) -> fp32 out, async-DMA staged
  gemm64x64_kernel<1><<<dim3(MROWS / 64, DIM / 64), 256, 0, stream>>>(
      O_bf, projw_bf, proj_b, out, nullptr, nullptr, MROWS, DIM, DIM);

  (void)in_sizes; (void)n_in; (void)out_size; (void)ws_size;
}
